// D_point_plus_seg_33706903339265
// MI455X (gfx1250) — compile-verified, matrix operand order checked by reading, not run on hardware
//
#include <hip/hip_runtime.h>
#include <hip/hip_bf16.h>

typedef __attribute__((ext_vector_type(16))) _Float16 v16h;
typedef __attribute__((ext_vector_type(8)))  _Float16 v8h;
typedef __attribute__((ext_vector_type(8)))  float    v8f;

// ---------------------------------------------------------------------------
// f16 fragment: elements 0..7 at p[0..8), elements 8..15 at p[16..24)
// (each chunk is one aligned 16-byte global_load_b128)
// ---------------------------------------------------------------------------
__device__ __forceinline__ v16h combine(v8h lo, v8h hi) {
  return __builtin_shufflevector(lo, hi, 0,1,2,3,4,5,6,7,8,9,10,11,12,13,14,15);
}

// ---------------------------------------------------------------------------
// GEMM via WMMA: Y[M x N](f16) = X[M x Kp](f16) @ Wp[N x Kp]^T(f16) + bias(f32)
// Kp multiple of 32; M multiple of 16.  One wave -> 16x64 strip, A reused 4x.
// All 10 loads per K-step issued before first use (MLP).
// ---------------------------------------------------------------------------
__global__ __launch_bounds__(256)
void gemm_wmma_kernel(const _Float16* __restrict__ X, const _Float16* __restrict__ Wp,
                      const float* __restrict__ bias, _Float16* __restrict__ Y,
                      int M, int N, int Kp) {
  const int wave   = blockIdx.x * (blockDim.x >> 5) + (threadIdx.x >> 5);
  const int strips = (N + 63) >> 6;
  const int tm = wave / strips;
  const int ts = wave % strips;
  if (tm * 16 >= M) return;                      // wave-uniform exit
  const int lane = threadIdx.x & 31;
  const int half = lane >> 4;
  const int mr   = lane & 15;

  const _Float16* Ar = X + (size_t)(tm * 16 + mr) * Kp + half * 8;
  const _Float16* B0;
  const _Float16* B1;
  const _Float16* B2;
  const _Float16* B3;
  {
    int n0 = ts * 64 + 0  + mr; n0 = (n0 < N) ? n0 : (N - 1);
    int n1 = ts * 64 + 16 + mr; n1 = (n1 < N) ? n1 : (N - 1);
    int n2 = ts * 64 + 32 + mr; n2 = (n2 < N) ? n2 : (N - 1);
    int n3 = ts * 64 + 48 + mr; n3 = (n3 < N) ? n3 : (N - 1);
    B0 = Wp + (size_t)n0 * Kp + half * 8;
    B1 = Wp + (size_t)n1 * Kp + half * 8;
    B2 = Wp + (size_t)n2 * Kp + half * 8;
    B3 = Wp + (size_t)n3 * Kp + half * 8;
  }

  v8f acc0 = {}, acc1 = {}, acc2 = {}, acc3 = {};
  for (int k0 = 0; k0 < Kp; k0 += 32) {
    // issue all loads first (distinct SSA values -> batched, partial waits)
    const v8h aL  = *(const v8h*)(Ar + k0);
    const v8h aH  = *(const v8h*)(Ar + k0 + 16);
    const v8h b0L = *(const v8h*)(B0 + k0);
    const v8h b0H = *(const v8h*)(B0 + k0 + 16);
    const v8h b1L = *(const v8h*)(B1 + k0);
    const v8h b1H = *(const v8h*)(B1 + k0 + 16);
    const v8h b2L = *(const v8h*)(B2 + k0);
    const v8h b2H = *(const v8h*)(B2 + k0 + 16);
    const v8h b3L = *(const v8h*)(B3 + k0);
    const v8h b3H = *(const v8h*)(B3 + k0 + 16);

    const v16h a = combine(aL, aH);
    acc0 = __builtin_amdgcn_wmma_f32_16x16x32_f16(false, a, false, combine(b0L, b0H), (short)0, acc0, false, false);
    acc1 = __builtin_amdgcn_wmma_f32_16x16x32_f16(false, a, false, combine(b1L, b1H), (short)0, acc1, false, false);
    acc2 = __builtin_amdgcn_wmma_f32_16x16x32_f16(false, a, false, combine(b2L, b2H), (short)0, acc2, false, false);
    acc3 = __builtin_amdgcn_wmma_f32_16x16x32_f16(false, a, false, combine(b3L, b3H), (short)0, acc3, false, false);
  }

  const int mbase = tm * 16 + 8 * half;
#pragma unroll
  for (int s = 0; s < 4; ++s) {
    const v8f acc = (s == 0) ? acc0 : (s == 1) ? acc1 : (s == 2) ? acc2 : acc3;
    const int n = ts * 64 + s * 16 + mr;
    if (n < N) {
      const float bv = bias ? bias[n] : 0.f;
#pragma unroll
      for (int v = 0; v < 8; ++v)
        Y[(size_t)(mbase + v) * N + n] = (_Float16)(acc[v] + bv);
    }
  }
}

// ---------------------------------------------------------------------------
// Pad+convert weight rows: Wp[n, 0..Kp)(f16) = {W[n, 0..K), 0...}
// ---------------------------------------------------------------------------
__global__ void pad_w16_kernel(const float* __restrict__ W, _Float16* __restrict__ Wp,
                               int N, int K, int Kp) {
  const size_t total = (size_t)N * Kp;
  const size_t stride = (size_t)gridDim.x * blockDim.x;
  for (size_t t = (size_t)blockIdx.x * blockDim.x + threadIdx.x; t < total; t += stride) {
    const int k = (int)(t % Kp);
    const size_t n = t / Kp;
    Wp[t] = (k < K) ? (_Float16)W[n * K + k] : (_Float16)0.f;
  }
}

// ---------------------------------------------------------------------------
// BatchNorm: per-channel mean/var over M rows (deterministic tree reduction)
// ---------------------------------------------------------------------------
__global__ __launch_bounds__(256)
void bn_stats_kernel(const _Float16* __restrict__ Y, float* __restrict__ mean,
                     float* __restrict__ var, int M, int N) {
  const int c = blockIdx.x;
  float s = 0.f, s2 = 0.f;
  for (int m = threadIdx.x; m < M; m += 256) {
    float v = (float)Y[(size_t)m * N + c];
    s += v; s2 += v * v;
  }
  __shared__ float sh[256], sh2[256];
  sh[threadIdx.x] = s; sh2[threadIdx.x] = s2;
  __syncthreads();
  for (int o = 128; o > 0; o >>= 1) {
    if (threadIdx.x < o) {
      sh[threadIdx.x]  += sh[threadIdx.x + o];
      sh2[threadIdx.x] += sh2[threadIdx.x + o];
    }
    __syncthreads();
  }
  if (threadIdx.x == 0) {
    float mu = sh[0] / (float)M;
    mean[c] = mu;
    var[c]  = sh2[0] / (float)M - mu * mu;
  }
}

__global__ void bn_apply_kernel(_Float16* __restrict__ Y, const float* __restrict__ mean,
                                const float* __restrict__ var, const float* __restrict__ g,
                                const float* __restrict__ be, size_t total, int N) {
  const size_t stride = (size_t)gridDim.x * blockDim.x;
  for (size_t t = (size_t)blockIdx.x * blockDim.x + threadIdx.x; t < total; t += stride) {
    const int c = (int)(t % N);
    float v = ((float)Y[t] - mean[c]) * rsqrtf(var[c] + 1e-5f) * g[c] + be[c];
    Y[t] = (_Float16)(v > 0.f ? v : 0.f);
  }
}

// ---------------------------------------------------------------------------
// Transpose (B,3,N) -> (B,N,3)  (geometry stays f32)
// ---------------------------------------------------------------------------
__global__ void transpose_in_kernel(const float* __restrict__ xyz, float* __restrict__ pts,
                                    int B, int N) {
  const size_t total = (size_t)B * N * 3;
  const size_t stride = (size_t)gridDim.x * blockDim.x;
  for (size_t t = (size_t)blockIdx.x * blockDim.x + threadIdx.x; t < total; t += stride) {
    const int c = (int)(t % 3);
    const size_t q = t / 3;
    const int n = (int)(q % N);
    const int b = (int)(q / N);
    pts[t] = xyz[((size_t)b * 3 + c) * N + n];
  }
}

// ---------------------------------------------------------------------------
// Farthest point sampling: one block per batch, point cloud in LDS.
// ---------------------------------------------------------------------------
__global__ __launch_bounds__(256)
void fps_kernel(const float* __restrict__ pts, int* __restrict__ fidx, int N, int S) {
  const int b = blockIdx.x;
  const float* P = pts + (size_t)b * N * 3;
  __shared__ float sx[4096], sy[4096], sz[4096], sdist[4096];
  __shared__ float rbest[256];
  __shared__ int   rbidx[256];
  __shared__ int   sfar;
  for (int i = threadIdx.x; i < N; i += 256) {
    sx[i] = P[i * 3 + 0]; sy[i] = P[i * 3 + 1]; sz[i] = P[i * 3 + 2];
    sdist[i] = 1e10f;
  }
  if (threadIdx.x == 0) sfar = 0;
  __syncthreads();
  for (int it = 0; it < S; ++it) {
    const int far = sfar;
    if (threadIdx.x == 0) fidx[b * S + it] = far;
    const float cx = sx[far], cy = sy[far], cz = sz[far];
    float best = -1.f; int bidx = 0;
    for (int i = threadIdx.x; i < N; i += 256) {
      const float dx = sx[i] - cx, dy = sy[i] - cy, dz = sz[i] - cz;
      const float d  = dx * dx + dy * dy + dz * dz;
      const float dm = fminf(sdist[i], d);
      sdist[i] = dm;
      if (dm > best) { best = dm; bidx = i; }
    }
    rbest[threadIdx.x] = best; rbidx[threadIdx.x] = bidx;
    __syncthreads();
    for (int o = 128; o > 0; o >>= 1) {
      if (threadIdx.x < o) {
        const float ob = rbest[threadIdx.x + o];
        const int   oi = rbidx[threadIdx.x + o];
        if (ob > rbest[threadIdx.x] ||
            (ob == rbest[threadIdx.x] && oi < rbidx[threadIdx.x])) {
          rbest[threadIdx.x] = ob; rbidx[threadIdx.x] = oi;
        }
      }
      __syncthreads();
    }
    if (threadIdx.x == 0) sfar = rbidx[0];
    __syncthreads();
  }
}

// ---------------------------------------------------------------------------
// Gather rows (f32): out[b,s,:] = in[b, idx[b,s], :]
// ---------------------------------------------------------------------------
__global__ void gather_rows_kernel(const float* __restrict__ in, const int* __restrict__ idx,
                                   float* __restrict__ out, int B, int Nin, int S, int C) {
  const size_t total = (size_t)B * S * C;
  const size_t stride = (size_t)gridDim.x * blockDim.x;
  for (size_t t = (size_t)blockIdx.x * blockDim.x + threadIdx.x; t < total; t += stride) {
    const int c = (int)(t % C);
    const size_t q = t / C;
    const int b = (int)(q / S);
    out[t] = in[((size_t)b * Nin + idx[q]) * C + c];
  }
}

// ---------------------------------------------------------------------------
// Ball query: one wave32 per query; ordered radius compaction via ballot.
// ---------------------------------------------------------------------------
__global__ __launch_bounds__(256)
void ball_query_kernel(const float* __restrict__ xyz, const float* __restrict__ new_xyz,
                       int* __restrict__ gidx, int B, int N, int S, int ns, float r2) {
  const int wavesPerBlock = blockDim.x >> 5;
  const int q = blockIdx.x * wavesPerBlock + (threadIdx.x >> 5);
  if (q >= B * S) return;
  const int b = q / S;
  const int lane = threadIdx.x & 31;
  const float* P = xyz + (size_t)b * N * 3;
  const float cx = new_xyz[(size_t)q * 3 + 0];
  const float cy = new_xyz[(size_t)q * 3 + 1];
  const float cz = new_xyz[(size_t)q * 3 + 2];
  int* out = gidx + (size_t)q * ns;
  int count = 0;
  unsigned firstIdx = 0u;
  for (int base = 0; base < N && count < ns; base += 32) {
    const int i = base + lane;
    bool inr = false;
    if (i < N) {
      const float dx = P[i * 3 + 0] - cx;
      const float dy = P[i * 3 + 1] - cy;
      const float dz = P[i * 3 + 2] - cz;
      inr = (dx * dx + dy * dy + dz * dz) <= r2;
    }
    const unsigned m32 = (unsigned)__ballot(inr);
    const int pos = count + __popc(m32 & ((1u << lane) - 1u));
    if (inr && pos < ns) out[pos] = i;
    if (inr && pos == 0) firstIdx = (unsigned)i;
    count += __popc(m32);
  }
  if (count > ns) count = ns;
  for (int off = 16; off > 0; off >>= 1) firstIdx |= __shfl_xor(firstIdx, off);
  const int first = (int)firstIdx;
  for (int j = lane; j < ns; j += 32)
    if (j >= count) out[j] = first;
}

// ---------------------------------------------------------------------------
// Grouping -> f16, padded stride Cpad:
//   out[u, 0..2] = xyz[idx]-center ; out[u, 3..3+Cp) = points[idx] ; pad = 0
// ---------------------------------------------------------------------------
__global__ void group_kernel(const float* __restrict__ xyz, const _Float16* __restrict__ points,
                             const float* __restrict__ new_xyz, const int* __restrict__ gidx,
                             _Float16* __restrict__ out, int B, int N, int S, int ns,
                             int Cp, int Cpad) {
  const size_t total = (size_t)B * S * ns * Cpad;
  const size_t stride = (size_t)gridDim.x * blockDim.x;
  for (size_t t = (size_t)blockIdx.x * blockDim.x + threadIdx.x; t < total; t += stride) {
    const int c = (int)(t % Cpad);
    const size_t u = t / Cpad;            // (b*S+s)*ns + j
    const size_t q = u / ns;              // b*S + s
    const int b = (int)(q / S);
    float v;
    if (c < 3) {
      const int p = gidx[u];
      v = xyz[((size_t)b * N + p) * 3 + c] - new_xyz[q * 3 + c];
    } else if (c < 3 + Cp) {
      const int p = gidx[u];
      v = (float)points[((size_t)b * N + p) * Cp + (c - 3)];
    } else {
      v = 0.f;
    }
    out[t] = (_Float16)v;
  }
}

// ---------------------------------------------------------------------------
// Max-pool over neighborhood axis: (Q, ns, C) -> (Q, C)  (f16)
// ---------------------------------------------------------------------------
__global__ void maxpool_kernel(const _Float16* __restrict__ X, _Float16* __restrict__ Y,
                               int Q, int ns, int C) {
  const size_t total = (size_t)Q * C;
  const size_t stride = (size_t)gridDim.x * blockDim.x;
  for (size_t t = (size_t)blockIdx.x * blockDim.x + threadIdx.x; t < total; t += stride) {
    const int c = (int)(t % C);
    const size_t q = t / C;
    float m = -3.0e38f;
    for (int j = 0; j < ns; ++j)
      m = fmaxf(m, (float)X[((size_t)q * ns + j) * C + c]);
    Y[t] = (_Float16)m;
  }
}

// ---------------------------------------------------------------------------
// Padded channel concat: out[q, :] = [A[q,0..C1), Bsrc[q/bDiv,0..C2), 0-pad]
// ---------------------------------------------------------------------------
template <typename TA, typename TB>
__global__ void concat2_kernel(const TA* __restrict__ A, const TB* __restrict__ Bs,
                               _Float16* __restrict__ out, int Q, int C1, int C2,
                               int bDiv, int Cpad) {
  const size_t total = (size_t)Q * Cpad;
  const size_t stride = (size_t)gridDim.x * blockDim.x;
  for (size_t t = (size_t)blockIdx.x * blockDim.x + threadIdx.x; t < total; t += stride) {
    const int c = (int)(t % Cpad);
    const size_t q = t / Cpad;
    float v;
    if (c < C1)            v = (float)A[q * C1 + c];
    else if (c < C1 + C2)  v = (float)Bs[(q / bDiv) * (size_t)C2 + (c - C1)];
    else                   v = 0.f;
    out[t] = (_Float16)v;
  }
}

// ---------------------------------------------------------------------------
// 3-NN search + inverse-distance weights (f32 geometry)
// ---------------------------------------------------------------------------
__global__ void knn3_kernel(const float* __restrict__ xyz1, const float* __restrict__ xyz2,
                            int* __restrict__ kidx, float* __restrict__ kw,
                            int B, int N1, int S) {
  const size_t total = (size_t)B * N1;
  const size_t stride = (size_t)gridDim.x * blockDim.x;
  for (size_t t = (size_t)blockIdx.x * blockDim.x + threadIdx.x; t < total; t += stride) {
    const int b = (int)(t / N1);
    const float px = xyz1[t * 3 + 0], py = xyz1[t * 3 + 1], pz = xyz1[t * 3 + 2];
    float d0 = 3.0e38f, d1 = 3.0e38f, d2 = 3.0e38f;
    int i0 = 0, i1 = 0, i2 = 0;
    const float* Q = xyz2 + (size_t)b * S * 3;
    for (int s = 0; s < S; ++s) {
      const float dx = px - Q[s * 3 + 0];
      const float dy = py - Q[s * 3 + 1];
      const float dz = pz - Q[s * 3 + 2];
      const float d = dx * dx + dy * dy + dz * dz;
      if (d < d0)      { d2 = d1; i2 = i1; d1 = d0; i1 = i0; d0 = d; i0 = s; }
      else if (d < d1) { d2 = d1; i2 = i1; d1 = d;  i1 = s; }
      else if (d < d2) { d2 = d;  i2 = s; }
    }
    float w0 = 1.f / (d0 + 1e-8f), w1 = 1.f / (d1 + 1e-8f), w2 = 1.f / (d2 + 1e-8f);
    const float wsum = w0 + w1 + w2;
    kidx[t * 3 + 0] = i0; kidx[t * 3 + 1] = i1; kidx[t * 3 + 2] = i2;
    kw[t * 3 + 0] = w0 / wsum; kw[t * 3 + 1] = w1 / wsum; kw[t * 3 + 2] = w2 / wsum;
  }
}

// ---------------------------------------------------------------------------
// out[q,:] = [points1[q, 0..C1) , sum_k w_k * points2[b, idx_k, 0..C2)]  (f16)
// ---------------------------------------------------------------------------
__global__ void interp_concat_kernel(const _Float16* __restrict__ P1, const _Float16* __restrict__ P2,
                                     const int* __restrict__ kidx, const float* __restrict__ kw,
                                     _Float16* __restrict__ out, int B, int N1, int S,
                                     int C1, int C2) {
  const int C = C1 + C2;
  const size_t total = (size_t)B * N1 * C;
  const size_t stride = (size_t)gridDim.x * blockDim.x;
  for (size_t t = (size_t)blockIdx.x * blockDim.x + threadIdx.x; t < total; t += stride) {
    const int c = (int)(t % C);
    const size_t q = t / C;
    const int b = (int)(q / N1);
    if (c < C1) {
      out[t] = P1[q * C1 + c];
    } else {
      const int cc = c - C1;
      float acc = 0.f;
#pragma unroll
      for (int k = 0; k < 3; ++k)
        acc += kw[q * 3 + k] * (float)P2[((size_t)b * S + kidx[q * 3 + k]) * C2 + cc];
      out[t] = (_Float16)acc;
    }
  }
}

// ---------------------------------------------------------------------------
// (B,N,O) f16 -> (B,O,N) f32 output
// ---------------------------------------------------------------------------
__global__ void transpose_out_kernel(const _Float16* __restrict__ Yin, float* __restrict__ out,
                                     int B, int N, int O) {
  const size_t total = (size_t)B * O * N;
  const size_t stride = (size_t)gridDim.x * blockDim.x;
  for (size_t t = (size_t)blockIdx.x * blockDim.x + threadIdx.x; t < total; t += stride) {
    const int n = (int)(t % N);
    const size_t r = t / N;
    const int o = (int)(r % O);
    const int b = (int)(r / O);
    out[t] = (float)Yin[((size_t)b * N + n) * O + o];
  }
}

// ===========================================================================
// Host orchestration
// ===========================================================================
namespace {
struct LayerP {
  const float *W, *b, *g, *be;
  int Cin, Cout, Kp;
  _Float16* Wp;
};
static inline int gb(size_t total, int bs) {
  size_t n = (total + bs - 1) / bs;
  if (n > 262144) n = 262144;
  return (int)n;
}
} // namespace

extern "C" void kernel_launch(void* const* d_in, const int* in_sizes, int n_in,
                              void* d_out, int out_size, void* d_ws, size_t ws_size,
                              hipStream_t stream) {
  (void)in_sizes; (void)n_in; (void)out_size; (void)ws_size;
  const int B = 16, N0 = 4096, S1 = 512, S2 = 128, NS = 64, OUTC = 50;

  int ip = 0;
  const float* xyz = (const float*)d_in[ip++];
  auto getLayer = [&](int cin, int cout) {
    LayerP L;
    L.W  = (const float*)d_in[ip++];
    L.b  = (const float*)d_in[ip++];
    L.g  = (const float*)d_in[ip++];
    L.be = (const float*)d_in[ip++];
    L.Cin = cin; L.Cout = cout;
    L.Kp = (cin + 31) & ~31;
    L.Wp = nullptr;
    return L;
  };
  LayerP sa1[3]; sa1[0]=getLayer(3,64);    sa1[1]=getLayer(64,64);   sa1[2]=getLayer(64,128);
  LayerP sa2[3]; sa2[0]=getLayer(131,128); sa2[1]=getLayer(128,128); sa2[2]=getLayer(128,256);
  LayerP sa3[3]; sa3[0]=getLayer(259,256); sa3[1]=getLayer(256,256); sa3[2]=getLayer(256,1024);
  LayerP fp3[2]; fp3[0]=getLayer(1280,256); fp3[1]=getLayer(256,256);
  LayerP fp2[2]; fp2[0]=getLayer(384,256);  fp2[1]=getLayer(256,128);
  LayerP fp1[3]; fp1[0]=getLayer(128,128);  fp1[1]=getLayer(128,128); fp1[2]=getLayer(128,128);
  LayerP head1 = getLayer(128,128);
  LayerP head2;
  head2.W = (const float*)d_in[ip++];
  head2.b = (const float*)d_in[ip++];
  head2.g = nullptr; head2.be = nullptr;
  head2.Cin = 128; head2.Cout = OUTC; head2.Kp = 128; head2.Wp = nullptr;

  // ---- workspace bump allocator (byte-based, 64B-aligned chunks) ----
  char* wb = (char*)d_ws;
  auto allocB = [&](size_t bytes) { char* p = wb; wb += (bytes + 63) & ~(size_t)63; return (void*)p; };
  float*     pts   = (float*)allocB((size_t)B * N0 * 3 * 4);
  int*       fidx1 = (int*)  allocB((size_t)B * S1 * 4);
  float*     nxyz1 = (float*)allocB((size_t)B * S1 * 3 * 4);
  int*       gidx1 = (int*)  allocB((size_t)B * S1 * NS * 4);
  int*       fidx2 = (int*)  allocB((size_t)B * S2 * 4);
  float*     nxyz2 = (float*)allocB((size_t)B * S2 * 3 * 4);
  int*       gidx2 = (int*)  allocB((size_t)B * S2 * NS * 4);
  _Float16*  l1p   = (_Float16*)allocB((size_t)B * S1 * 128 * 2);
  _Float16*  l2p   = (_Float16*)allocB((size_t)B * S2 * 256 * 2);
  _Float16*  l3p   = (_Float16*)allocB((size_t)B * 1024 * 2);
  float*     meanb = (float*)allocB(1024 * 4);
  float*     varb  = (float*)allocB(1024 * 4);
  int*       kidx  = (int*)  allocB((size_t)B * N0 * 3 * 4);
  float*     kw    = (float*)allocB((size_t)B * N0 * 3 * 4);
  const size_t BIG = (size_t)B * S1 * NS * 128;           // elements
  _Float16*  bufA  = (_Float16*)allocB(BIG * 2);
  _Float16*  bufB  = (_Float16*)allocB(BIG * 2);

  // ---- pad+convert all weights to f16 (tiny, once per launch, graph-safe) ----
  LayerP* layers[18] = { &sa1[0], &sa1[1], &sa1[2], &sa2[0], &sa2[1], &sa2[2],
                         &sa3[0], &sa3[1], &sa3[2], &fp3[0], &fp3[1],
                         &fp2[0], &fp2[1], &fp1[0], &fp1[1], &fp1[2],
                         &head1, &head2 };
  for (int i = 0; i < 18; ++i) {
    LayerP* L = layers[i];
    const size_t n = (size_t)L->Cout * L->Kp;
    L->Wp = (_Float16*)allocB(n * 2);
    pad_w16_kernel<<<gb(n, 256), 256, 0, stream>>>(L->W, L->Wp, L->Cout, L->Cin, L->Kp);
  }

  auto run_gemm = [&](const _Float16* X, const LayerP& L, _Float16* Y, int M) {
    const size_t waves = (size_t)(M / 16) * ((L.Cout + 63) / 64);
    const int blocks = (int)((waves + 7) / 8);
    gemm_wmma_kernel<<<blocks, 256, 0, stream>>>(X, L.Wp, L.b, Y, M, L.Cout, L.Kp);
  };
  auto run_bn = [&](_Float16* Y, const LayerP& L, int M) {
    bn_stats_kernel<<<L.Cout, 256, 0, stream>>>(Y, meanb, varb, M, L.Cout);
    const size_t tot = (size_t)M * L.Cout;
    bn_apply_kernel<<<gb(tot, 256), 256, 0, stream>>>(Y, meanb, varb, L.g, L.be, tot, L.Cout);
  };
  auto run_layer = [&](const _Float16* X, const LayerP& L, _Float16* Y, int M) {
    run_gemm(X, L, Y, M);
    run_bn(Y, L, M);
  };

  // ---- input transpose (B,3,N) -> (B,N,3) ----
  transpose_in_kernel<<<gb((size_t)B * N0 * 3, 256), 256, 0, stream>>>(xyz, pts, B, N0);

  // ================= SA1 =================
  fps_kernel<<<B, 256, 0, stream>>>(pts, fidx1, N0, S1);
  gather_rows_kernel<<<gb((size_t)B * S1 * 3, 256), 256, 0, stream>>>(pts, fidx1, nxyz1, B, N0, S1, 3);
  ball_query_kernel<<<(B * S1 + 7) / 8, 256, 0, stream>>>(pts, nxyz1, gidx1, B, N0, S1, NS, 0.015f * 0.015f);
  group_kernel<<<gb((size_t)B * S1 * NS * 32, 256), 256, 0, stream>>>(pts, nullptr, nxyz1, gidx1, bufA, B, N0, S1, NS, 0, 32);
  const int M1 = B * S1 * NS;
  run_layer(bufA, sa1[0], bufB, M1);
  run_layer(bufB, sa1[1], bufA, M1);
  run_layer(bufA, sa1[2], bufB, M1);
  maxpool_kernel<<<gb((size_t)B * S1 * 128, 256), 256, 0, stream>>>(bufB, l1p, B * S1, NS, 128);

  // ================= SA2 =================
  fps_kernel<<<B, 256, 0, stream>>>(nxyz1, fidx2, S1, S2);
  gather_rows_kernel<<<gb((size_t)B * S2 * 3, 256), 256, 0, stream>>>(nxyz1, fidx2, nxyz2, B, S1, S2, 3);
  ball_query_kernel<<<(B * S2 + 7) / 8, 256, 0, stream>>>(nxyz1, nxyz2, gidx2, B, S1, S2, NS, 0.04f * 0.04f);
  group_kernel<<<gb((size_t)B * S2 * NS * 160, 256), 256, 0, stream>>>(nxyz1, l1p, nxyz2, gidx2, bufA, B, S1, S2, NS, 128, 160);
  const int M2 = B * S2 * NS;
  run_layer(bufA, sa2[0], bufB, M2);
  run_layer(bufB, sa2[1], bufA, M2);
  run_layer(bufA, sa2[2], bufB, M2);
  maxpool_kernel<<<gb((size_t)B * S2 * 256, 256), 256, 0, stream>>>(bufB, l2p, B * S2, NS, 256);

  // ================= SA3 (group-all) =================
  const int M3 = B * S2;  // 2048 rows
  concat2_kernel<float, _Float16><<<gb((size_t)M3 * 288, 256), 256, 0, stream>>>(nxyz2, l2p, bufA, M3, 3, 256, 1, 288);
  run_layer(bufA, sa3[0], bufB, M3);
  run_layer(bufB, sa3[1], bufA, M3);
  run_layer(bufA, sa3[2], bufB, M3);
  maxpool_kernel<<<gb((size_t)B * 1024, 256), 256, 0, stream>>>(bufB, l3p, B, S2, 1024);

  // ================= FP3 (broadcast global feature) =================
  concat2_kernel<_Float16, _Float16><<<gb((size_t)M3 * 1280, 256), 256, 0, stream>>>(l2p, l3p, bufA, M3, 256, 1024, S2, 1280);
  run_layer(bufA, fp3[0], bufB, M3);
  run_layer(bufB, fp3[1], l2p, M3);          // l2p <- fp3 output (B,128,256)

  // ================= FP2 =================
  knn3_kernel<<<gb((size_t)B * S1, 256), 256, 0, stream>>>(nxyz1, nxyz2, kidx, kw, B, S1, S2);
  interp_concat_kernel<<<gb((size_t)B * S1 * 384, 256), 256, 0, stream>>>(l1p, l2p, kidx, kw, bufA, B, S1, S2, 128, 256);
  const int M4 = B * S1;  // 8192 rows
  run_layer(bufA, fp2[0], bufB, M4);
  run_layer(bufB, fp2[1], l1p, M4);          // l1p <- fp2 output (B,512,128)

  // ================= FP1 =================
  knn3_kernel<<<gb((size_t)B * N0, 256), 256, 0, stream>>>(pts, nxyz1, kidx, kw, B, N0, S1);
  interp_concat_kernel<<<gb((size_t)B * N0 * 128, 256), 256, 0, stream>>>(nullptr, l1p, kidx, kw, bufA, B, N0, S1, 0, 128);
  const int M0 = B * N0;  // 65536 rows
  run_layer(bufA, fp1[0], bufB, M0);
  run_layer(bufB, fp1[1], bufA, M0);
  run_layer(bufA, fp1[2], bufB, M0);         // l0 features in bufB

  // ================= Head =================
  run_layer(bufB, head1, bufA, M0);          // conv1 + BN + relu
  run_gemm(bufA, head2, bufB, M0);           // conv2 (no BN), N=50
  transpose_out_kernel<<<gb((size_t)B * OUTC * N0, 256), 256, 0, stream>>>(bufB, (float*)d_out, B, N0, OUTC);
}
